// LSTM_20066087207035
// MI455X (gfx1250) — compile-verified, matrix operand order checked by reading, not run on hardware
//
#include <hip/hip_runtime.h>
#include <hip/hip_bf16.h>

typedef __bf16 bf16;
typedef __attribute__((ext_vector_type(16))) __bf16 v16bf;
typedef __attribute__((ext_vector_type(8)))  __bf16 v8bf;
typedef __attribute__((ext_vector_type(8)))  float   v8f;

#define BB   256          // batch
#define TT   512          // time steps
#define HH   512          // hidden
#define NG   2048         // 4*H gate columns
#define KMAX 1024         // layer-1 fused K ([h1_t | h_prev])
#define SW   (KMAX + 8)   // LDS row stride (elements) — avoids bank conflicts
#define XSW  528          // x-tile LDS row stride
#define NWG  256          // persistent workgroups (16 M-groups x 16 N-groups)

// ---- workspace layout (bytes) ----
#define OFF_BAR   0
#define OFF_HPING 256                                   // bf16 [2][B*H]
#define OFF_HFIN  (OFF_HPING + 2*BB*HH*2)               // f32  [B*H]
#define OFF_W0    (OFF_HFIN  + BB*HH*4)                 // bf16 [NG][H]
#define OFF_W1    (OFF_W0    + (size_t)NG*HH*2)         // bf16 [NG][KMAX]
#define OFF_H1    (OFF_W1    + (size_t)NG*KMAX*2)       // bf16 [T][B*H]

#define SM_W   (128*SW*2)
#define SM_A   (16*SW*2)
#define SM_G   (4*16*32*4)
#define SM_X   (16*XSW*2)
#define SMEM_BYTES (SM_W + SM_A + SM_G + SM_X)          // 322,304 B < 327,680

union ABf { v16bf v; v8bf h[2]; };

#if __has_builtin(__builtin_amdgcn_tanhf)
__device__ __forceinline__ float fast_tanh(float x) { return __builtin_amdgcn_tanhf(x); }
#else
__device__ __forceinline__ float fast_tanh(float x) { return tanhf(x); }
#endif
__device__ __forceinline__ float fast_sigmoid(float x) {
    return 0.5f * (1.0f + fast_tanh(0.5f * x));   // v_tanh_f32 on CDNA5
}

__device__ __forceinline__ void grid_barrier(unsigned* bar) {
    __syncthreads();
    if (threadIdx.x == 0) {
        __threadfence();                               // release h stores device-wide
        asm volatile("s_wait_storecnt 0" ::: "memory");
        unsigned gen = __hip_atomic_load(&bar[1], __ATOMIC_RELAXED, __HIP_MEMORY_SCOPE_AGENT);
        unsigned a = __hip_atomic_fetch_add(&bar[0], 1u, __ATOMIC_ACQ_REL, __HIP_MEMORY_SCOPE_AGENT);
        if (a == NWG - 1u) {
            __hip_atomic_store(&bar[0], 0u, __ATOMIC_RELAXED, __HIP_MEMORY_SCOPE_AGENT);
            __hip_atomic_fetch_add(&bar[1], 1u, __ATOMIC_RELEASE, __HIP_MEMORY_SCOPE_AGENT);
        } else {
            while (__hip_atomic_load(&bar[1], __ATOMIC_ACQUIRE, __HIP_MEMORY_SCOPE_AGENT) == gen)
                __builtin_amdgcn_s_sleep(2);
        }
        __threadfence();                               // acquire: invalidate stale WGP$ lines
    }
    __syncthreads();
}

// -------- prep: bf16-pack weights, zero barrier --------
__global__ __launch_bounds__(256)
void lstm_prep(const float* __restrict__ whh0, const float* __restrict__ wih1,
               const float* __restrict__ whh1, char* __restrict__ ws) {
    size_t idx = (size_t)blockIdx.x * 256 + threadIdx.x;
    if (idx == 0) { ((unsigned*)ws)[0] = 0u; ((unsigned*)ws)[1] = 0u; }
    bf16* W0 = (bf16*)(ws + OFF_W0);
    bf16* W1 = (bf16*)(ws + OFF_W1);
    const size_t n0 = (size_t)NG * HH;
    const size_t n1 = (size_t)NG * KMAX;
    if (idx < n0) {
        W0[idx] = (bf16)whh0[idx];
    } else if (idx < n0 + n1) {
        size_t j = idx - n0;
        size_t row = j >> 10, k = j & 1023;
        float v = (k < 512) ? wih1[row * 512 + k] : whh1[row * 512 + (k - 512)];
        W1[j] = (bf16)v;
    }
}

// -------- persistent 2-layer LSTM --------
__global__ __launch_bounds__(256)
void lstm_persistent(const float* __restrict__ x,     // [B][T] (I==1)
                     const float* __restrict__ wih0,  // [NG]
                     char* __restrict__ ws) {
    extern __shared__ char smem[];
    bf16*  Wl   = (bf16*)smem;                        // [128][SW]
    bf16*  Al   = (bf16*)(smem + SM_W);               // [16][SW]
    float* gbuf = (float*)(smem + SM_W + SM_A);       // [4][16][32]
    bf16*  Xl   = (bf16*)(smem + SM_W + SM_A + SM_G); // [16][XSW]

    unsigned* bar   = (unsigned*)ws;
    bf16*  hping    = (bf16*)(ws + OFF_HPING);
    float* hfin     = (float*)(ws + OFF_HFIN);
    const bf16* W0g = (const bf16*)(ws + OFF_W0);
    const bf16* W1g = (const bf16*)(ws + OFF_W1);
    bf16*  h1seq    = (bf16*)(ws + OFF_H1);

    const int tid  = threadIdx.x;
    const int mg = blockIdx.x >> 4, ng = blockIdx.x & 15;
    const int m0 = mg * 16, c0 = ng * 32;
    const int wave = tid >> 5, lane = tid & 31;
    const int gw = wave >> 1, half = wave & 1;        // gate id, half of 32-col slice
    const int nloc = lane & 15, hi = lane >> 4;
    const int hi8 = hi * 8;
    const int brow = gw * 32 + half * 16 + nloc;      // B row in local weight slab
    const int e0 = tid * 2;                           // 2 cells per thread

    float creg[2];

    for (int layer = 0; layer < 2; ++layer) {
        const int Keff = layer ? KMAX : HH;
        const bf16* Wg = layer ? W1g : W0g;
        const int kg = Keff;
        // ---- load this WG's 128 gate rows into LDS (once per layer) ----
        for (int r = 0; r < 128; ++r) {
            const int g = r >> 5, j = r & 31;
            const uint4* src = (const uint4*)(Wg + (size_t)(g * HH + c0 + j) * kg);
            uint4* dst = (uint4*)(Wl + (size_t)r * SW);
            for (int c = tid; c < kg / 8; c += 256) dst[c] = src[c];
        }
        if (layer == 0) {   // stage this WG's x tile once: [16 rows][T] -> bf16 LDS
            for (int r = 0; r < 16; ++r) {
                const float* src = x + (size_t)(m0 + r) * TT;
                for (int c = tid; c < TT; c += 256) Xl[r * XSW + c] = (bf16)src[c];
            }
        }
        creg[0] = creg[1] = 0.f;
        __syncthreads();

        for (int t = 0; t < TT; ++t) {
            const int kend = (t > 0) ? Keff : (layer ? 512 : 0);
            // ---- stage A tile: [h1_t | h_prev] (layer1) or h_prev (layer0) ----
            if (layer) {
                const bf16* h1t = h1seq + (size_t)t * BB * HH;
                for (int r = 0; r < 16; ++r) {
                    const uint4* src = (const uint4*)(h1t + (size_t)(m0 + r) * HH);
                    uint4* dst = (uint4*)(Al + (size_t)r * SW);
                    for (int c = tid; c < HH / 8; c += 256) dst[c] = src[c];
                }
                if (t + 1 < TT) {   // prefetch next step's input rows (global_prefetch_b8)
                    const char* p = (const char*)(h1seq + (size_t)(t + 1) * BB * HH
                                     + (size_t)(m0 + (tid & 15)) * HH) + (size_t)(tid >> 4) * 64;
                    __builtin_prefetch(p, 0, 1);
                }
            }
            if (t > 0) {
                const bf16* hprev = hping + (size_t)((t - 1) & 1) * BB * HH;
                const int koff = layer ? 512 : 0;
                for (int r = 0; r < 16; ++r) {
                    const uint4* src = (const uint4*)(hprev + (size_t)(m0 + r) * HH);
                    uint4* dst = (uint4*)(Al + (size_t)r * SW + koff);
                    for (int c = tid; c < HH / 8; c += 256) dst[c] = src[c];
                }
            }
            __syncthreads();

            // ---- accumulate gates tile: 16x16, K = kend via WMMA bf16 ----
            v8f acc;
            if (layer == 0) {   // input term: x[b,t] * w_ih_l0[gate_col] (I==1)
                const float wv = wih0[gw * HH + c0 + half * 16 + nloc];
                #pragma unroll
                for (int v = 0; v < 8; ++v)
                    acc[v] = (float)Xl[(v + 8 * hi) * XSW + t] * wv;
            } else {
                #pragma unroll
                for (int v = 0; v < 8; ++v) acc[v] = 0.f;
            }
            const bf16* arow = Al + (size_t)nloc * SW;
            const bf16* brw  = Wl + (size_t)brow * SW;
            const int nblk = kend >> 7;              // blocks of 128 K (0, 4, or 8)
            for (int blk = 0; blk < nblk; ++blk) {
                const int kb = blk * 128;
                ABf ua[4], ub[4];
                #pragma unroll
                for (int u = 0; u < 4; ++u) {        // issue all DS loads first
                    const int k = kb + u * 32;
                    ua[u].h[0] = *(const v8bf*)(arow + k + hi8);
                    ua[u].h[1] = *(const v8bf*)(arow + k + hi8 + 16);
                    ub[u].h[0] = *(const v8bf*)(brw  + k + hi8);
                    ub[u].h[1] = *(const v8bf*)(brw  + k + hi8 + 16);
                }
                #pragma unroll
                for (int u = 0; u < 4; ++u)          // then 4 back-to-back WMMAs
                    acc = __builtin_amdgcn_wmma_f32_16x16x32_bf16(
                            false, ua[u].v, false, ub[u].v, (short)0, acc, false, false);
            }

            // ---- exchange gates across waves via LDS ----
            #pragma unroll
            for (int v = 0; v < 8; ++v)
                gbuf[(gw * 16 + v + 8 * hi) * 32 + half * 16 + nloc] = acc[v];
            __syncthreads();

            // ---- LSTM cell update: each thread owns 2 (b, h) cells ----
            bf16* hdst = hping + (size_t)(t & 1) * BB * HH;
            #pragma unroll
            for (int e = 0; e < 2; ++e) {
                const int idx = e0 + e;
                const int i = idx >> 5, j = idx & 31;
                const float gi = gbuf[(0 * 16 + i) * 32 + j];
                const float gf = gbuf[(1 * 16 + i) * 32 + j];
                const float gg = gbuf[(2 * 16 + i) * 32 + j];
                const float go = gbuf[(3 * 16 + i) * 32 + j];
                const float c  = fast_sigmoid(gf) * creg[e]
                               + fast_sigmoid(gi) * fast_tanh(gg);
                creg[e] = c;
                const float h = fast_sigmoid(go) * fast_tanh(c);
                const size_t off = (size_t)(m0 + i) * HH + c0 + j;
                hdst[off] = (bf16)h;
                if (layer == 0) h1seq[(size_t)t * BB * HH + off] = (bf16)h;
                else if (t == TT - 1) hfin[off] = h;
            }
            grid_barrier(bar);
        }
    }
}

// -------- LayerNorm + 512->1 head --------
__global__ __launch_bounds__(256)
void ln_head(const float* __restrict__ hfin, const float* __restrict__ gamma,
             const float* __restrict__ beta, const float* __restrict__ wout,
             float* __restrict__ out) {
    __shared__ float red[256];
    const int b = blockIdx.x, tid = threadIdx.x;
    const float* row = hfin + (size_t)b * HH;
    const float x0 = row[tid], x1 = row[tid + 256];
    red[tid] = x0 + x1; __syncthreads();
    for (int s = 128; s > 0; s >>= 1) { if (tid < s) red[tid] += red[tid + s]; __syncthreads(); }
    const float mu = red[0] * (1.f / 512.f); __syncthreads();
    const float d0 = x0 - mu, d1 = x1 - mu;
    red[tid] = d0 * d0 + d1 * d1; __syncthreads();
    for (int s = 128; s > 0; s >>= 1) { if (tid < s) red[tid] += red[tid + s]; __syncthreads(); }
    const float rs = rsqrtf(red[0] * (1.f / 512.f) + 1e-5f); __syncthreads();
    const float n0 = d0 * rs * gamma[tid] + beta[tid];
    const float n1 = d1 * rs * gamma[tid + 256] + beta[tid + 256];
    red[tid] = n0 * wout[tid] + n1 * wout[tid + 256]; __syncthreads();
    for (int s = 128; s > 0; s >>= 1) { if (tid < s) red[tid] += red[tid + s]; __syncthreads(); }
    if (tid == 0) out[b] = red[0];
}

extern "C" void kernel_launch(void* const* d_in, const int* in_sizes, int n_in,
                              void* d_out, int out_size, void* d_ws, size_t ws_size,
                              hipStream_t stream) {
    (void)in_sizes; (void)n_in; (void)out_size; (void)ws_size;
    const float* x     = (const float*)d_in[0];
    const float* wih0  = (const float*)d_in[1];
    const float* whh0  = (const float*)d_in[2];
    const float* wih1  = (const float*)d_in[3];
    const float* whh1  = (const float*)d_in[4];
    const float* gamma = (const float*)d_in[5];
    const float* beta  = (const float*)d_in[6];
    const float* wout  = (const float*)d_in[7];
    char* ws = (char*)d_ws;

    static bool attr_set = false;   // idempotent, deterministic
    if (!attr_set) {
        hipFuncSetAttribute((const void*)lstm_persistent,
                            hipFuncAttributeMaxDynamicSharedMemorySize, SMEM_BYTES);
        attr_set = true;
    }

    const size_t prep_elems = (size_t)NG * HH + (size_t)NG * KMAX;
    const int prep_blocks = (int)((prep_elems + 255) / 256);
    lstm_prep<<<prep_blocks, 256, 0, stream>>>(whh0, wih1, whh1, ws);
    lstm_persistent<<<NWG, 256, SMEM_BYTES, stream>>>(x, wih0, ws);
    ln_head<<<BB, 256, 0, stream>>>((const float*)(ws + OFF_HFIN), gamma, beta, wout,
                                    (float*)d_out);
}
